// GCN_dgl_20186346291610
// MI455X (gfx1250) — compile-verified, matrix operand order checked by reading, not run on hardware
//
#include <hip/hip_runtime.h>
#include <hip/hip_bf16.h>

typedef __attribute__((ext_vector_type(16))) __bf16 v16bf;
typedef __attribute__((ext_vector_type(8)))  float  v8f;

__device__ __forceinline__ unsigned bfbits(float f) {   // fp32 -> bf16 bits (RNE)
    union { float f; unsigned u; } v; v.f = f;
    return (v.u + 0x7FFFu + ((v.u >> 16) & 1u)) >> 16;
}

// ---------------------------------------------------------------------------
// Dense GEMM  C[M,NOUT] = A[M,K] @ Wt[NOUT,K]^T  (bf16 operands, fp32 out)
// A row-major [M][K] bf16; Wt column-major-of-W = row-major [NOUT][K] bf16.
// One wave -> one 16x16 tile; inner loop = 4x b128 loads + 1 wmma.
// Requires M % 16 == 0 (true for this problem: M = 100000).
// ---------------------------------------------------------------------------
template <int K, int NOUT>
__global__ __launch_bounds__(32 * (NOUT / 16))
void gemm_bf16_wmma(const __bf16* __restrict__ A, const __bf16* __restrict__ Wt,
                    float* __restrict__ C) {
    const int lane = threadIdx.x & 31;
    const int wave = threadIdx.x >> 5;
    const int rm   = blockIdx.x * 16;            // row-tile base
    const int cn   = wave * 16;                  // col-tile base
    const int mr   = lane & 15;
    const int koff = (lane >> 4) * 8;            // K-half select per ISA layout

    const __bf16* arow = A  + (size_t)(rm + mr) * K + koff;  // lane holds row
    const __bf16* brow = Wt + (size_t)(cn + mr) * K + koff;  // lane holds col of W

    v8f acc = {};
    for (int kb = 0; kb < K; kb += 32) {
        union { v16bf v; uint4 q[2]; } a, b;
        a.q[0] = *(const uint4*)(arow + kb);          // K = kb+koff   .. +7
        a.q[1] = *(const uint4*)(arow + kb + 16);     // K = kb+koff+16.. +23
        b.q[0] = *(const uint4*)(brow + kb);
        b.q[1] = *(const uint4*)(brow + kb + 16);
        acc = __builtin_amdgcn_wmma_f32_16x16x32_bf16(
            /*neg_a=*/false, a.v, /*neg_b=*/false, b.v,
            /*c_mod=*/(short)0, acc, /*reuse_a=*/false, /*reuse_b=*/false);
    }
    const int rbase = rm + ((lane >> 4) << 3);   // D: VGPR i -> row rbase+i
    float* crow = C + (size_t)rbase * NOUT + cn + (lane & 15);
    #pragma unroll
    for (int i = 0; i < 8; ++i) crow[(size_t)i * NOUT] = acc[i];
}

// ---------------------------------------------------------------------------
// Conversions
// ---------------------------------------------------------------------------
__global__ void k_f32_to_bf16_pk(const float* __restrict__ src,
                                 unsigned* __restrict__ dst, size_t npairs) {
    size_t i = (size_t)blockIdx.x * blockDim.x + threadIdx.x;
    if (i < npairs)
        dst[i] = bfbits(src[2 * i]) | (bfbits(src[2 * i + 1]) << 16);
}

// W [K][NOUT] fp32 -> Wt [NOUT][K] bf16
__global__ void k_wt_bf16(const float* __restrict__ W,
                          unsigned short* __restrict__ Wt, int K, int NOUT) {
    int i = blockIdx.x * blockDim.x + threadIdx.x;
    if (i < K * NOUT) {
        int k = i / NOUT, n = i % NOUT;
        Wt[(size_t)n * K + k] = (unsigned short)bfbits(W[i]);
    }
}

// ---------------------------------------------------------------------------
// Degrees / norms
// ---------------------------------------------------------------------------
__global__ void k_fill0(float* __restrict__ p, size_t n) {
    size_t i = (size_t)blockIdx.x * blockDim.x + threadIdx.x;
    if (i < n) p[i] = 0.0f;
}

__global__ void k_degrees(const long long* __restrict__ src,
                          const long long* __restrict__ dst,
                          unsigned* __restrict__ deg_out,
                          unsigned* __restrict__ deg_in, int E) {
    int e = blockIdx.x * blockDim.x + threadIdx.x;
    if (e < E) {
        atomicAdd(&deg_out[(int)src[e]], 1u);
        atomicAdd(&deg_in[(int)dst[e]], 1u);
    }
}

__global__ void k_norms(unsigned* __restrict__ deg, int n) {  // in-place uint -> float
    int i = blockIdx.x * blockDim.x + threadIdx.x;
    if (i < n) {
        unsigned d = deg[i];
        float f = rsqrtf(d > 0 ? (float)d : 1.0f);
        ((float*)deg)[i] = f;
    }
}

// ---------------------------------------------------------------------------
// SpMM: agg[dst] += h[src] * n_src[src]   (edge-parallel, fp32 L2 atomics)
// ---------------------------------------------------------------------------
template <int F, int EPB>
__global__ __launch_bounds__(F)
void k_spmm(const float* __restrict__ h, const long long* __restrict__ src,
            const long long* __restrict__ dst, const float* __restrict__ nsrc,
            float* __restrict__ agg, int E) {
    const int t  = threadIdx.x;
    const int e0 = blockIdx.x * EPB;
    #pragma unroll
    for (int i = 0; i < EPB; ++i) {
        int e = e0 + i;
        if (e >= E) return;
        int s = (int)src[e];
        int d = (int)dst[e];
        float val = h[(size_t)s * F + t] * nsrc[s];
        unsafeAtomicAdd(&agg[(size_t)d * F + t], val);   // global_atomic_add_f32
    }
}

// out_bf16[i,c] = bf16(agg[i,c]*n_dst[i] + b[c])  -- feeds only the next GEMM
template <int F>
__global__ void k_scale_bias_bf16(const float* __restrict__ agg,
                                  const float* __restrict__ ndst,
                                  const float* __restrict__ b,
                                  unsigned* __restrict__ out, int n) {
    int i = blockIdx.x * blockDim.x + threadIdx.x;   // over n*(F/2) pairs
    if (i < n * (F / 2)) {
        int r  = i / (F / 2);
        int c2 = (i % (F / 2)) * 2;
        float nd = ndst[r];
        unsigned lo = bfbits(agg[(size_t)r * F + c2]     * nd + b[c2]);
        unsigned hi = bfbits(agg[(size_t)r * F + c2 + 1] * nd + b[c2 + 1]);
        out[i] = lo | (hi << 16);
    }
}

// fused scale + bias + row softmax for the final 64-wide layer; 1 wave / row
__global__ void k_softmax64(const float* __restrict__ agg,
                            const float* __restrict__ ndst,
                            const float* __restrict__ b,
                            float* __restrict__ out, int n) {
    const int lane = threadIdx.x & 31;
    const int row  = blockIdx.x * (blockDim.x >> 5) + (threadIdx.x >> 5);
    if (row >= n) return;
    float nd = ndst[row];
    float v0 = agg[(size_t)row * 64 + lane]      * nd + b[lane];
    float v1 = agg[(size_t)row * 64 + 32 + lane] * nd + b[32 + lane];
    float m = fmaxf(v0, v1);
    #pragma unroll
    for (int off = 16; off > 0; off >>= 1) m = fmaxf(m, __shfl_xor(m, off, 32));
    float e0 = __expf(v0 - m), e1 = __expf(v1 - m);
    float s = e0 + e1;
    #pragma unroll
    for (int off = 16; off > 0; off >>= 1) s += __shfl_xor(s, off, 32);
    float inv = 1.0f / s;
    out[(size_t)row * 64 + lane]      = e0 * inv;
    out[(size_t)row * 64 + 32 + lane] = e1 * inv;
}

// ---------------------------------------------------------------------------
extern "C" void kernel_launch(void* const* d_in, const int* in_sizes, int n_in,
                              void* d_out, int out_size, void* d_ws, size_t ws_size,
                              hipStream_t stream) {
    const float*     x   = (const float*)d_in[0];
    const long long* src = (const long long*)d_in[1];   // int64 in reference
    const long long* dst = (const long long*)d_in[2];
    const float* W0 = (const float*)d_in[3];
    const float* b0 = (const float*)d_in[4];
    const float* W1 = (const float*)d_in[5];
    const float* b1 = (const float*)d_in[6];
    const float* W2 = (const float*)d_in[7];
    const float* b2 = (const float*)d_in[8];

    const int NFEAT = 256, NHID = 128, NCLASS = 64;
    const int n = in_sizes[0] / NFEAT;   // 100000 (multiple of 16)
    const int E = in_sizes[1];           // 3200000

    // workspace layout (all offsets 16B-aligned):
    // [nsrc n f32][ndst n f32][bufA n*128 f32][bufB n*128 f32]
    // [act16 n*256 bf16][wt0 256*128 bf16][wt1 128*128 bf16][wt2 128*64 bf16]
    float*  ws    = (float*)d_ws;
    float*  nsrc  = ws;
    float*  ndst  = ws + n;
    float*  bufA  = ws + 2 * (size_t)n;
    float*  bufB  = bufA + (size_t)n * NHID;
    __bf16* act16 = (__bf16*)(bufB + (size_t)n * NHID);
    __bf16* wt0   = act16 + (size_t)n * NFEAT;
    __bf16* wt1   = wt0 + 256 * 128;
    __bf16* wt2   = wt1 + 128 * 128;

    const int TPB = 256;
    auto blocks = [](size_t work, int tpb) { return (unsigned)((work + tpb - 1) / tpb); };

    // --- degree norms -------------------------------------------------------
    k_fill0<<<blocks(2 * (size_t)n, TPB), TPB, 0, stream>>>(ws, 2 * (size_t)n);
    k_degrees<<<blocks(E, TPB), TPB, 0, stream>>>(src, dst, (unsigned*)nsrc, (unsigned*)ndst, E);
    k_norms<<<blocks(2 * (size_t)n, TPB), TPB, 0, stream>>>((unsigned*)ws, 2 * n);

    // --- one-time bf16 operand prep ----------------------------------------
    k_f32_to_bf16_pk<<<blocks((size_t)n * NFEAT / 2, TPB), TPB, 0, stream>>>(
        x, (unsigned*)act16, (size_t)n * NFEAT / 2);
    k_wt_bf16<<<blocks(256 * 128, TPB), TPB, 0, stream>>>(W0, (unsigned short*)wt0, 256, 128);
    k_wt_bf16<<<blocks(128 * 128, TPB), TPB, 0, stream>>>(W1, (unsigned short*)wt1, 128, 128);
    k_wt_bf16<<<blocks(128 * 64,  TPB), TPB, 0, stream>>>(W2, (unsigned short*)wt2, 128, 64);

    const unsigned gM = (unsigned)(n / 16);   // exact row tiles

    // --- layer 0: 256 -> 128 ------------------------------------------------
    gemm_bf16_wmma<256, 128><<<gM, 256, 0, stream>>>(act16, wt0, bufA);
    k_fill0<<<blocks((size_t)n * 128, TPB), TPB, 0, stream>>>(bufB, (size_t)n * 128);
    k_spmm<128, 4><<<blocks(E, 4), 128, 0, stream>>>(bufA, src, dst, nsrc, bufB, E);
    k_scale_bias_bf16<128><<<blocks((size_t)n * 64, TPB), TPB, 0, stream>>>(
        bufB, ndst, b0, (unsigned*)act16, n);

    // --- layer 1: 128 -> 128 ------------------------------------------------
    gemm_bf16_wmma<128, 128><<<gM, 256, 0, stream>>>(act16, wt1, bufA);
    k_fill0<<<blocks((size_t)n * 128, TPB), TPB, 0, stream>>>(bufB, (size_t)n * 128);
    k_spmm<128, 4><<<blocks(E, 4), 128, 0, stream>>>(bufA, src, dst, nsrc, bufB, E);
    k_scale_bias_bf16<128><<<blocks((size_t)n * 64, TPB), TPB, 0, stream>>>(
        bufB, ndst, b1, (unsigned*)act16, n);

    // --- layer 2: 128 -> 64 + softmax --------------------------------------
    gemm_bf16_wmma<128, 64><<<gM, 128, 0, stream>>>(act16, wt2, bufA);
    k_fill0<<<blocks((size_t)n * 64, TPB), TPB, 0, stream>>>(bufB, (size_t)n * 64);
    k_spmm<64, 4><<<blocks(E, 4), 64, 0, stream>>>(bufA, src, dst, nsrc, bufB, E);
    k_softmax64<<<blocks(n, 8), 256, 0, stream>>>(bufB, ndst, b2, (float*)d_out, n);
}
